// CrossAttention_77309411328210
// MI455X (gfx1250) — matrix operand order repaired and checked by reading, not compiled, not run on hardware
//
#include <hip/hip_runtime.h>
#include <hip/hip_bf16.h>

#define TOK_N  4096
#define DMODEL 512
#define NHEAD  8
#define DHEAD  64
#define BATCH  2
#define MROWS  (BATCH * TOK_N)   // 8192

typedef __attribute__((ext_vector_type(8)))  _Float16 v8h;
typedef __attribute__((ext_vector_type(16))) _Float16 v16h;
typedef __attribute__((ext_vector_type(8)))  float    v8f;

__device__ __forceinline__ v16h cat8(v8h lo, v8h hi) {
  return __builtin_shufflevector(lo, hi, 0,1,2,3,4,5,6,7,8,9,10,11,12,13,14,15);
}
__device__ __forceinline__ v8f zero8() {
  v8f z = {0.f,0.f,0.f,0.f,0.f,0.f,0.f,0.f};
  return z;
}
__device__ __forceinline__ v8f wmma16(v16h a, v16h b, v8f c) {
  return __builtin_amdgcn_wmma_f32_16x16x32_f16(
      false, a, false, b, (short)0, c, false, false);
}

// CDNA5 async global->LDS copy (16B per lane), tracked by ASYNCcnt.
// vdst = LDS byte offset (low 32 bits of generic LDS pointer, per aperture rule).
__device__ __forceinline__ void async_cp16(unsigned lds_off, const _Float16* g) {
  asm volatile("global_load_async_to_lds_b128 %0, %1, off"
               :: "v"(lds_off), "v"(g) : "memory");
}
#define ASYNC_WAIT_8() asm volatile("s_wait_asynccnt 0x8" ::: "memory")
#define ASYNC_WAIT_0() asm volatile("s_wait_asynccnt 0x0" ::: "memory")

// ---------------- conversion kernels ----------------

__global__ void cvt_f32_to_f16(const float* __restrict__ in,
                               _Float16* __restrict__ out, int n) {
  int i = (blockIdx.x * blockDim.x + threadIdx.x) * 4;
  if (i + 3 < n) {
    float4 f = *(const float4*)(in + i);
    out[i + 0] = (_Float16)f.x;
    out[i + 1] = (_Float16)f.y;
    out[i + 2] = (_Float16)f.z;
    out[i + 3] = (_Float16)f.w;
  }
}

// Wt[n][k] = W[k][n], f32 -> f16, 512x512
__global__ void cvt_transpose_w(const float* __restrict__ W,
                                _Float16* __restrict__ Wt) {
  int i = blockIdx.x * blockDim.x + threadIdx.x;
  int nrow = i >> 9;
  int kcol = i & 511;
  Wt[i] = (_Float16)W[kcol * DMODEL + nrow];
}

// ---------------- WMMA GEMM: 128x64 block tile, 8 waves, 32x32 per wave ------
// C[M=8192,N=512] = A[M,512] x Bt[N,512]^T
// MODE 0: f16 natural; MODE 1: f16 V-transpose [B,H,DH,TOK_N]; MODE 2: f32+bias

template <int MODE>
__global__ __launch_bounds__(256)
void gemm_wmma(const _Float16* __restrict__ A, const _Float16* __restrict__ Bt,
               const float* __restrict__ bias, _Float16* __restrict__ out16,
               float* __restrict__ outf) {
  __shared__ _Float16 As[2][128][40];
  __shared__ _Float16 Bs[2][64][40];
  const int n0   = blockIdx.x * 64;
  const int m0   = blockIdx.y * 128;
  const int tid  = threadIdx.x;
  const int lane = tid & 31;
  const int wave = tid >> 5;
  const int lrow = lane & 15;
  const int hi   = (lane & 16) ? 1 : 0;
  const int koff = hi * 8;    // A frag: lanes>=16 hold K 8..15 / 24..31
  const int klo  = hi * 16;   // unused; B frag uses koff too: lanes>=16 hold K 8..15 / 24..31
  const int wrow = (wave >> 1) * 32;
  const int wcol = (wave & 1) * 32;

  // loader mapping: A 128x32 halves (16/thread), B 64x32 halves (8/thread)
  const int ar = tid >> 1, aseg = (tid & 1) * 16;
  const int br = tid >> 2, bseg = (tid & 3) * 8;

  v8f acc[2][2];
  #pragma unroll
  for (int i = 0; i < 2; ++i)
    #pragma unroll
    for (int j = 0; j < 2; ++j) acc[i][j] = zero8();

  // prologue: stage k0=0 into buffer 0
  {
    const _Float16* ga = A  + (size_t)(m0 + ar) * DMODEL + aseg;
    const _Float16* gb = Bt + (size_t)(n0 + br) * DMODEL + bseg;
    *(v8h*)&As[0][ar][aseg]     = *(const v8h*)ga;
    *(v8h*)&As[0][ar][aseg + 8] = *(const v8h*)(ga + 8);
    *(v8h*)&Bs[0][br][bseg]     = *(const v8h*)gb;
  }
  __syncthreads();

  for (int k0 = 0; k0 < DMODEL; k0 += 32) {
    const int cur = (k0 >> 5) & 1;
    const int nxt = cur ^ 1;
    const bool have_next = (k0 + 32) < DMODEL;

    // issue next tile's global loads early (overlap with WMMA)
    v8h pa0, pa1, pb0;
    if (have_next) {
      const _Float16* ga = A  + (size_t)(m0 + ar) * DMODEL + k0 + 32 + aseg;
      const _Float16* gb = Bt + (size_t)(n0 + br) * DMODEL + k0 + 32 + bseg;
      pa0 = *(const v8h*)ga;
      pa1 = *(const v8h*)(ga + 8);
      pb0 = *(const v8h*)gb;
    }

    // compute 32x32 per wave: 2 A frags x 2 B frags -> 4 WMMA
    v16h af[2], bf[2];
    #pragma unroll
    for (int i = 0; i < 2; ++i) {
      const int arow = wrow + i * 16 + lrow;
      af[i] = cat8(*(const v8h*)&As[cur][arow][koff],
                   *(const v8h*)&As[cur][arow][16 + koff]);
    }
    #pragma unroll
    for (int j = 0; j < 2; ++j) {
      const int brow = wcol + j * 16 + lrow;
      bf[j] = cat8(*(const v8h*)&Bs[cur][brow][koff],
                   *(const v8h*)&Bs[cur][brow][16 + koff]);
    }
    #pragma unroll
    for (int i = 0; i < 2; ++i)
      #pragma unroll
      for (int j = 0; j < 2; ++j)
        acc[i][j] = wmma16(af[i], bf[j], acc[i][j]);

    if (have_next) {
      *(v8h*)&As[nxt][ar][aseg]     = pa0;
      *(v8h*)&As[nxt][ar][aseg + 8] = pa1;
      *(v8h*)&Bs[nxt][br][bseg]     = pb0;
    }
    __syncthreads();
  }

  // straight-line epilogue (MODE resolved at compile time)
  #pragma unroll
  for (int i = 0; i < 2; ++i) {
    #pragma unroll
    for (int j = 0; j < 2; ++j) {
      const int mrow0 = m0 + wrow + i * 16 + hi * 8;
      const int cg    = n0 + wcol + j * 16 + lrow;
      #pragma unroll
      for (int r = 0; r < 8; ++r) {
        float v = acc[i][j][r];
        int mg = mrow0 + r;
        if (MODE == 0) {
          out16[(size_t)mg * DMODEL + cg] = (_Float16)v;
        } else if (MODE == 1) {
          int bidx = mg >> 12, nt = mg & (TOK_N - 1);
          int h = cg >> 6, dh = cg & 63;
          out16[(size_t)((bidx * NHEAD + h) * DHEAD + dh) * TOK_N + nt] = (_Float16)v;
        } else {
          outf[(size_t)mg * DMODEL + cg] = v + bias[cg];
        }
      }
    }
  }
}

// ---------------- flash attention with async-LDS double buffering ------------
// grid: (TOK_N/64, NHEAD, BATCH), 128 threads (4 waves, 16 query rows each)

__global__ __launch_bounds__(128)
void attn_wmma(const _Float16* __restrict__ Q16, const _Float16* __restrict__ K16,
               const _Float16* __restrict__ Vt16, _Float16* __restrict__ O16) {
  __shared__ _Float16 Qs[64][72];
  __shared__ _Float16 Ks[2][64][72];
  __shared__ _Float16 Vs[2][64][72];   // [dh][key]
  __shared__ _Float16 Ps[64][72];      // [query][key]
  const int qt = blockIdx.x;
  const int h  = blockIdx.y;
  const int b  = blockIdx.z;
  const int q0 = qt * 64;
  const int tid = threadIdx.x, lane = tid & 31, wave = tid >> 5;
  const int lrow = lane & 15;
  const int hi   = (lane & 16) ? 1 : 0;
  const int koff = hi * 8, klo = hi * 16;
  const int ldr = tid >> 1, ldseg = (tid & 1) * 32;

  {  // stage Q tile (sync copies)
    const _Float16* gq = Q16 + (size_t)(b * TOK_N + q0 + ldr) * DMODEL + h * DHEAD + ldseg;
    #pragma unroll
    for (int i = 0; i < 4; ++i)
      *(v8h*)&Qs[ldr][ldseg + i * 8] = *(const v8h*)(gq + i * 8);
  }

  // async staging of one K/V chunk into LDS buffer `buf` (8 b128 per thread)
  auto issue_chunk = [&](int c, int buf) {
    const _Float16* gk = K16  + (size_t)(b * TOK_N + c * 64 + ldr) * DMODEL + h * DHEAD + ldseg;
    const _Float16* gv = Vt16 + (size_t)((b * NHEAD + h) * DHEAD + ldr) * TOK_N + c * 64 + ldseg;
    unsigned lk = (unsigned)(size_t)&Ks[buf][ldr][ldseg];
    unsigned lv = (unsigned)(size_t)&Vs[buf][ldr][ldseg];
    #pragma unroll
    for (int i = 0; i < 4; ++i) {
      async_cp16(lk + i * 16, gk + i * 8);
      async_cp16(lv + i * 16, gv + i * 8);
    }
  };

  const int NC = TOK_N / 64;
  issue_chunk(0, 0);
  issue_chunk(1, 1);

  float m_run[8], l_run[8];
  v8f oacc[4];
  #pragma unroll
  for (int r = 0; r < 8; ++r) { m_run[r] = -1e30f; l_run[r] = 0.f; }
  #pragma unroll
  for (int j = 0; j < 4; ++j) oacc[j] = zero8();

  for (int c = 0; c < NC; ++c) {
    const int cur = c & 1;
    // ASYNCcnt retires in order: <=8 outstanding means chunk c's 8 ops done.
    if (c + 1 < NC) { ASYNC_WAIT_8(); } else { ASYNC_WAIT_0(); }
    __syncthreads();

    // S = Q * K^T  (16 queries x 64 keys per wave), f32 accum
    v8f sacc[4];
    #pragma unroll
    for (int j = 0; j < 4; ++j) sacc[j] = zero8();
    #pragma unroll
    for (int ks = 0; ks < 64; ks += 32) {
      const int arow = wave * 16 + lrow;
      v16h afrag = cat8(*(const v8h*)&Qs[arow][ks + koff],
                        *(const v8h*)&Qs[arow][ks + 16 + koff]);
      #pragma unroll
      for (int j = 0; j < 4; ++j) {
        const int brow = j * 16 + lrow;
        v16h bfrag = cat8(*(const v8h*)&Ks[cur][brow][ks + koff],
                          *(const v8h*)&Ks[cur][brow][ks + 16 + koff]);
        sacc[j] = wmma16(afrag, bfrag, sacc[j]);
      }
    }

    // online softmax per row; reductions stay inside each 16-lane half
    #pragma unroll
    for (int r = 0; r < 8; ++r) {
      float mx = -1e30f;
      #pragma unroll
      for (int j = 0; j < 4; ++j) {
        float s = sacc[j][r] * 0.125f;  // DH^-0.5
        sacc[j][r] = s;
        mx = fmaxf(mx, s);
      }
      #pragma unroll
      for (int msk = 1; msk < 16; msk <<= 1)
        mx = fmaxf(mx, __shfl_xor(mx, msk, 32));
      float mn   = fmaxf(m_run[r], mx);
      float corr = __expf(m_run[r] - mn);
      float rs   = 0.f;
      const int prow = wave * 16 + r + hi * 8;
      #pragma unroll
      for (int j = 0; j < 4; ++j) {
        float p = __expf(sacc[j][r] - mn);
        rs += p;
        Ps[prow][j * 16 + lrow] = (_Float16)p;
      }
      #pragma unroll
      for (int msk = 1; msk < 16; msk <<= 1)
        rs += __shfl_xor(rs, msk, 32);
      l_run[r] = l_run[r] * corr + rs;
      m_run[r] = mn;
      #pragma unroll
      for (int j = 0; j < 4; ++j) oacc[j][r] *= corr;
    }
    __syncthreads();

    // O += P * V
    #pragma unroll
    for (int ks = 0; ks < 64; ks += 32) {
      const int arow = wave * 16 + lrow;
      v16h afrag = cat8(*(const v8h*)&Ps[arow][ks + koff],
                        *(const v8h*)&Ps[arow][ks + 16 + koff]);
      #pragma unroll
      for (int j = 0; j < 4; ++j) {
        const int brow = j * 16 + lrow;
        v16h bfrag = cat8(*(const v8h*)&Vs[cur][brow][ks + koff],
                          *(const v8h*)&Vs[cur][brow][ks + 16 + koff]);
        oacc[j] = wmma16(afrag, bfrag, oacc[j]);
      }
    }
    __syncthreads();  // all waves done with buffer `cur` before refilling it

    if (c + 2 < NC) issue_chunk(c + 2, cur);
  }

  #pragma unroll
  for (int r = 0; r < 8; ++r) {
    float inv = 1.f / l_run[r];
    int row = b * TOK_N + q0 + wave * 16 + r + hi * 8;
    #pragma unroll
    for (int j = 0; j < 4; ++j)
      O16[(size_t)row * DMODEL + h * DHEAD + j * 16 + lrow] =
          (_Float16)(oacc[j][r] * inv);
  }
}

// ---------------- launch ----------------

extern "C" void kernel_launch(void* const* d_in, const int* in_sizes, int n_in,
                              void* d_out, int out_size, void* d_ws, size_t ws_size,
                              hipStream_t stream) {
  const float* x  = (const float*)d_in[0];
  const float* Wq = (const float*)d_in[1];
  const float* Wk = (const float*)d_in[2];
  const float* Wv = (const float*)d_in[3];
  const float* Wo = (const float*)d_in[4];
  const float* bo = (const float*)d_in[5];
  float* out = (float*)d_out;

  _Float16* ws = (_Float16*)d_ws;
  const size_t XSZ = (size_t)MROWS * DMODEL;
  const size_t WSZ = (size_t)DMODEL * DMODEL;
  _Float16* x16  = ws;
  _Float16* WqT  = x16  + XSZ;
  _Float16* WkT  = WqT  + WSZ;
  _Float16* WvT  = WkT  + WSZ;
  _Float16* WoT  = WvT  + WSZ;
  _Float16* Q16  = WoT  + WSZ;
  _Float16* K16  = Q16  + XSZ;
  _Float16* Vt16 = K16  + XSZ;
  _Float16* O16  = Vt16 + XSZ;

  cvt_f32_to_f16<<<(int)(XSZ / 4 / 256), 256, 0, stream>>>(x, x16, (int)XSZ);
  cvt_transpose_w<<<(int)(WSZ / 256), 256, 0, stream>>>(Wq, WqT);
  cvt_transpose_w<<<(int)(WSZ / 256), 256, 0, stream>>>(Wk, WkT);
  cvt_transpose_w<<<(int)(WSZ / 256), 256, 0, stream>>>(Wv, WvT);
  cvt_transpose_w<<<(int)(WSZ / 256), 256, 0, stream>>>(Wo, WoT);

  dim3 gg(DMODEL / 64, MROWS / 128);  // 8 x 64 blocks, 256 threads
  gemm_wmma<0><<<gg, 256, 0, stream>>>(x16, WqT, nullptr, Q16,  nullptr);
  gemm_wmma<0><<<gg, 256, 0, stream>>>(x16, WkT, nullptr, K16,  nullptr);
  gemm_wmma<1><<<gg, 256, 0, stream>>>(x16, WvT, nullptr, Vt16, nullptr);

  dim3 ga(TOK_N / 64, NHEAD, BATCH);
  attn_wmma<<<ga, 128, 0, stream>>>(Q16, K16, Vt16, O16);

  gemm_wmma<2><<<gg, 256, 0, stream>>>(O16, WoT, bo, nullptr, out);
}
